// PointwiseConvNet3D_RNN_21698174780258
// MI455X (gfx1250) — matrix operand order repaired and checked by reading, not compiled, not run on hardware
//
#include <hip/hip_runtime.h>

// ---------------------------------------------------------------------------
// PointwiseConvNet3D_RNN on MI455X (gfx1250, wave32, WMMA)
//   16384 voxels x (4-layer GRU, H=64, T=24) x (BN+SiLU MLP)
//   1 wave = 16 voxels = one WMMA M-tile; all GEMMs -> v_wmma_f32_16x16x32_f16
// ---------------------------------------------------------------------------

typedef __attribute__((ext_vector_type(16))) _Float16 v16h;
typedef __attribute__((ext_vector_type(8)))  _Float16 v8h;
typedef __attribute__((ext_vector_type(8)))  float    v8f;

#define SHFL16 0,1,2,3,4,5,6,7,8,9,10,11,12,13,14,15

__device__ __forceinline__ v8f wmma16(v16h a, v16h b, v8f c) {
  // (neg_a, A, neg_b, B, c_mod, C, reuse_a, reuse_b)
  return __builtin_amdgcn_wmma_f32_16x16x32_f16(false, a, false, b, (short)0, c,
                                                false, false);
}

// A fragment (16x32 f16): lane holds row m=lane&15; lanes 0-15 get K=0..7,16..23,
// lanes 16-31 get K=8..15,24..31 (per ISA 16-bit A layout). `row` points at the
// 64-wide f16 row, kc selects K chunk (0: K=0..31, 1: K=32..63).
__device__ __forceinline__ v16h fragA(const _Float16* row, int kc, int hi) {
  const v8h* p = (const v8h*)(row + kc * 32);
  v8h a0 = p[hi];      // halves kc*32 + 8*hi
  v8h a1 = p[hi + 2];  // halves kc*32 + 16 + 8*hi
  return __builtin_shufflevector(a0, a1, SHFL16);
}

// B fragment (32x16 f16): lane holds column n=lane&15; lanes 0-15 K=0..7,16..23,
// lanes 16-31 K=8..15,24..31. `row` = weight row (output unit) of 64 contiguous f16.
__device__ __forceinline__ v16h fragB(const _Float16* row, int kc, int hi) {
  const v8h* p = (const v8h*)(row + kc * 32);
  return __builtin_shufflevector(p[hi], p[hi + 2], SHFL16);
}

__device__ __forceinline__ float sigm(float x) {
  return 1.0f / (1.0f + __expf(-x));
}
__device__ __forceinline__ float tanh_f(float x) {
  x = fminf(fmaxf(x, -15.0f), 15.0f);
  float e = __expf(2.0f * x);
  return (e - 1.0f) / (e + 1.0f);
}
__device__ __forceinline__ float silu(float x) { return x * sigm(x); }

// LDS layout (bytes):
//   whh f16 [192][64]           : 24576
//   wih f16 [192][64]           : 24576   (whh+wih overlaid by fcw[5][64][64])
//   wih0 f32[192]               :   768
//   bsum f32[128]               :   512   (bih+bhh for r,z gates)
//   bihn f32[64], bhhn f32[64]  :   512
//   v    f32[4][16][24]         :  6144
//   out  f16[4][24][16][64]     : 196608
static constexpr int SMEM_BYTES = 253696;

__global__ void __launch_bounds__(128, 1)
gru_voxel_wmma(const float* __restrict__ xg_in, const int* __restrict__ lengths,
               const float* __restrict__ sf, const float* __restrict__ bp,
               const float* __restrict__ wih0g, const float* __restrict__ whh0g,
               const float* __restrict__ bih0g, const float* __restrict__ bhh0g,
               const float* __restrict__ wihLg, const float* __restrict__ whhLg,
               const float* __restrict__ bihLg, const float* __restrict__ bhhLg,
               const float* __restrict__ bng, const float* __restrict__ bnb,
               const float* __restrict__ bnm, const float* __restrict__ bnv,
               const float* __restrict__ fcWg, const float* __restrict__ fcbg,
               const float* __restrict__ fcWoutg, const float* __restrict__ fcboutg,
               float* __restrict__ outG) {
  extern __shared__ char smem[];
  _Float16* lds_whh = (_Float16*)smem;           // 192*64
  _Float16* lds_wih = lds_whh + 192 * 64;        // 192*64
  _Float16* lds_fcw = lds_whh;                   // 5*64*64 overlay (20480 <= 24576 halves x2)
  float* lds_wih0 = (float*)(lds_wih + 192 * 64);
  float* lds_bsum = lds_wih0 + 192;
  float* lds_bihn = lds_bsum + 128;
  float* lds_bhhn = lds_bihn + 64;
  float* lds_v    = lds_bhhn + 64;               // 4*16*24
  _Float16* lds_out = (_Float16*)(lds_v + 4 * 16 * 24);  // 4*24*16*64

  const int tid  = threadIdx.x;
  const int lane = tid & 31;
  const int wave = tid >> 5;
  const int n16  = lane & 15;   // tile column / A-row index
  const int hi   = lane >> 4;   // half-wave selector

  float*    vW   = lds_v + wave * (16 * 24);
  _Float16* outW = lds_out + wave * (24 * 16 * 64);

  const int voxBase = blockIdx.x * 64 + wave * 16;  // first voxel of this wave
  const int nb = voxBase >> 12;                     // batch index (4096 voxels each)
  const int sp = voxBase & 4095;                    // spatial offset

  // ---- load 16 voxel sequences, apply scale/bias (v = x*s[t%3] + b[t%3]) ----
#pragma unroll 1
  for (int it = 0; it < 12; it++) {
    int idx = it * 32 + lane;            // 0..383 -> (m, t)
    int mm = idx / 24, tt = idx - mm * 24;
    float val = xg_in[(size_t)nb * 98304 + (size_t)tt * 4096 + sp + mm];
    int j = tt % 3;
    vW[mm * 24 + tt] = val * sf[j] + bp[j];
  }

  // =========================== 4 stacked GRU layers ===========================
#pragma unroll 1
  for (int L = 0; L < 4; L++) {
    __syncthreads();
    // cooperative f32->f16 weight staging
    const float* Wh = (L == 0) ? whh0g : whhLg + (size_t)(L - 1) * (192 * 64);
#pragma unroll 1
    for (int idx = tid; idx < 192 * 64; idx += 128) lds_whh[idx] = (_Float16)Wh[idx];
    if (L > 0) {
      const float* Wx = wihLg + (size_t)(L - 1) * (192 * 64);
#pragma unroll 1
      for (int idx = tid; idx < 192 * 64; idx += 128) lds_wih[idx] = (_Float16)Wx[idx];
    } else {
#pragma unroll 1
      for (int idx = tid; idx < 192; idx += 128) lds_wih0[idx] = wih0g[idx];
    }
    const float* bi = (L == 0) ? bih0g : bihLg + (L - 1) * 192;
    const float* bh = (L == 0) ? bhh0g : bhhLg + (L - 1) * 192;
    if (tid < 128) lds_bsum[tid] = bi[tid] + bh[tid];
    if (tid < 64) { lds_bihn[tid] = bi[128 + tid]; lds_bhhn[tid] = bh[128 + tid]; }
    __syncthreads();

    // h state in accumulator layout: hreg[i][r] = h[m = r+8*hi][c = 16*i+n16]
    v8f hreg[4];
#pragma unroll
    for (int i = 0; i < 4; i++) {
#pragma unroll
      for (int r = 0; r < 8; r++) hreg[i][r] = 0.0f;
    }

#pragma unroll 1
    for (int t = 0; t < 24; t++) {
      v16h xa0, xa1, ha0, ha1;
      if (L > 0) {  // input GEMM reads previous layer's out[t] (pre-overwrite)
        const _Float16* xrow = outW + (t * 16 + n16) * 64;
        xa0 = fragA(xrow, 0, hi); xa1 = fragA(xrow, 1, hi);
      }
      if (t > 0) {  // recurrent GEMM reads out[t-1] written by this layer
        const _Float16* hrow = outW + ((t - 1) * 16 + n16) * 64;
        ha0 = fragA(hrow, 0, hi); ha1 = fragA(hrow, 1, hi);
      }
      float vv[8];
      if (L == 0) {
#pragma unroll
        for (int r = 0; r < 8; r++) vv[r] = vW[(r + 8 * hi) * 24 + t];
      } else {
#pragma unroll
        for (int r = 0; r < 8; r++) vv[r] = 0.0f;
      }

      // accumulators: srz[0..3]=r-gate sum, srz[4..7]=z-gate sum, xnn/hnn = n-gate
      v8f srz[8], xnn[4], hnn[4];
#pragma unroll
      for (int i = 0; i < 8; i++) {
        float b  = lds_bsum[16 * i + n16];
        float w0 = (L == 0) ? lds_wih0[16 * i + n16] : 0.0f;
#pragma unroll
        for (int r = 0; r < 8; r++) srz[i][r] = b + w0 * vv[r];
      }
#pragma unroll
      for (int i = 0; i < 4; i++) {
        float bin = lds_bihn[16 * i + n16];
        float bhn = lds_bhhn[16 * i + n16];
        float w0  = (L == 0) ? lds_wih0[128 + 16 * i + n16] : 0.0f;
#pragma unroll
        for (int r = 0; r < 8; r++) { xnn[i][r] = bin + w0 * vv[r]; hnn[i][r] = bhn; }
      }

      // 12 N-tiles x 2 K-chunks x (x-GEMM + h-GEMM)
#pragma unroll
      for (int i = 0; i < 12; i++) {
        if (L > 0) {
          const _Float16* brow = lds_wih + (16 * i + n16) * 64;
          v16h b0 = fragB(brow, 0, hi), b1 = fragB(brow, 1, hi);
          if (i < 8) { srz[i] = wmma16(xa0, b0, srz[i]); srz[i] = wmma16(xa1, b1, srz[i]); }
          else       { xnn[i-8] = wmma16(xa0, b0, xnn[i-8]); xnn[i-8] = wmma16(xa1, b1, xnn[i-8]); }
        }
        if (t > 0) {
          const _Float16* brow = lds_whh + (16 * i + n16) * 64;
          v16h b0 = fragB(brow, 0, hi), b1 = fragB(brow, 1, hi);
          if (i < 8) { srz[i] = wmma16(ha0, b0, srz[i]); srz[i] = wmma16(ha1, b1, srz[i]); }
          else       { hnn[i-8] = wmma16(ha0, b0, hnn[i-8]); hnn[i-8] = wmma16(ha1, b1, hnn[i-8]); }
        }
      }

      // gate math + h update, store f16 copy to out[t]
#pragma unroll
      for (int i = 0; i < 4; i++) {
#pragma unroll
        for (int r = 0; r < 8; r++) {
          float rg = sigm(srz[i][r]);
          float zg = sigm(srz[4 + i][r]);
          float ng = tanh_f(xnn[i][r] + rg * hnn[i][r]);
          float hv = (1.0f - zg) * ng + zg * hreg[i][r];
          hreg[i][r] = hv;
          outW[(t * 16 + r + 8 * hi) * 64 + 16 * i + n16] = (_Float16)hv;
        }
      }
      asm volatile("s_wait_dscnt 0x0" ::: "memory");  // LDS store->load fence (same wave)
    }
  }

  // ================================ MLP head =================================
  __syncthreads();
#pragma unroll 1
  for (int idx = tid; idx < 5 * 64 * 64; idx += 128) lds_fcw[idx] = (_Float16)fcWg[idx];
  __syncthreads();

  const int len = lengths[nb];  // uniform across the wave's 16 voxels
  const int l1 = len - 1, l4 = len - 4;

  v8f hreg[4];
#pragma unroll
  for (int i = 0; i < 4; i++) {
    int c = 16 * i + n16;
#pragma unroll
    for (int r = 0; r < 8; r++) {
      int m = r + 8 * hi;
      hreg[i][r] = (float)outW[(l1 * 16 + m) * 64 + c] +
                   (float)outW[(l4 * 16 + m) * 64 + c];
    }
  }

  _Float16* ldsH = outW;  // reuse out[0] region as [16][64] staging tile
#pragma unroll 1
  for (int layer = 0; layer < 5; layer++) {
    // BN + SiLU (accumulator layout), stage f16 for the next GEMM
#pragma unroll
    for (int i = 0; i < 4; i++) {
      int c = layer * 64 + 16 * i + n16;
      float gs = bng[c] / sqrtf(bnv[c] + 1e-5f);
      float mu = bnm[c], be = bnb[c];
#pragma unroll
      for (int r = 0; r < 8; r++) {
        float u = gs * (hreg[i][r] - mu) + be;
        ldsH[(r + 8 * hi) * 64 + 16 * i + n16] = (_Float16)silu(u);
      }
    }
    asm volatile("s_wait_dscnt 0x0" ::: "memory");
    v16h a0 = fragA(ldsH + n16 * 64, 0, hi);
    v16h a1 = fragA(ldsH + n16 * 64, 1, hi);
    v8f nh[4];
#pragma unroll
    for (int nt = 0; nt < 4; nt++) {
      float bb = fcbg[layer * 64 + 16 * nt + n16];
      v8f acc;
#pragma unroll
      for (int r = 0; r < 8; r++) acc[r] = bb;
      const _Float16* brow = lds_fcw + (layer * 64 + 16 * nt + n16) * 64;
      acc = wmma16(a0, fragB(brow, 0, hi), acc);
      acc = wmma16(a1, fragB(brow, 1, hi), acc);
      nh[nt] = acc;
    }
#pragma unroll
    for (int i = 0; i < 4; i++) hreg[i] = nh[i];
  }

  // final BN+SiLU, 64->1 projection via VALU + cross-lane reduction
  float part[8];
#pragma unroll
  for (int r = 0; r < 8; r++) part[r] = 0.0f;
#pragma unroll
  for (int i = 0; i < 4; i++) {
    int c  = 16 * i + n16;
    int c5 = 5 * 64 + c;
    float gs = bng[c5] / sqrtf(bnv[c5] + 1e-5f);
    float mu = bnm[c5], be = bnb[c5];
    float wo = fcWoutg[c];
#pragma unroll
    for (int r = 0; r < 8; r++) {
      float u = gs * (hreg[i][r] - mu) + be;
      part[r] += silu(u) * wo;
    }
  }
#pragma unroll
  for (int off = 1; off < 16; off <<= 1) {
#pragma unroll
    for (int r = 0; r < 8; r++) part[r] += __shfl_xor(part[r], off, 32);
  }
  if (n16 == 0) {
    float bo = fcboutg[0];
#pragma unroll
    for (int r = 0; r < 8; r++) outG[voxBase + r + 8 * hi] = part[r] + bo;
  }
}

extern "C" void kernel_launch(void* const* d_in, const int* in_sizes, int n_in,
                              void* d_out, int out_size, void* d_ws, size_t ws_size,
                              hipStream_t stream) {
  (void)in_sizes; (void)n_in; (void)out_size; (void)d_ws; (void)ws_size;
  const float* x        = (const float*)d_in[0];
  const int*   lengths  = (const int*)d_in[1];
  const float* sf       = (const float*)d_in[2];
  const float* bp       = (const float*)d_in[3];
  const float* wih0     = (const float*)d_in[4];
  const float* whh0     = (const float*)d_in[5];
  const float* bih0     = (const float*)d_in[6];
  const float* bhh0     = (const float*)d_in[7];
  const float* wihL     = (const float*)d_in[8];
  const float* whhL     = (const float*)d_in[9];
  const float* bihL     = (const float*)d_in[10];
  const float* bhhL     = (const float*)d_in[11];
  const float* bng      = (const float*)d_in[12];
  const float* bnb      = (const float*)d_in[13];
  const float* bnm      = (const float*)d_in[14];
  const float* bnv      = (const float*)d_in[15];
  const float* fcW      = (const float*)d_in[16];
  const float* fcb      = (const float*)d_in[17];
  const float* fcWout   = (const float*)d_in[18];
  const float* fcbout   = (const float*)d_in[19];
  float* out = (float*)d_out;

  (void)hipFuncSetAttribute((const void*)gru_voxel_wmma,
                            hipFuncAttributeMaxDynamicSharedMemorySize, SMEM_BYTES);

  // 16384 voxels / (4 waves x 16 voxels) = 256 blocks of 128 threads
  gru_voxel_wmma<<<dim3(256), dim3(128), SMEM_BYTES, stream>>>(
      x, lengths, sf, bp, wih0, whh0, bih0, bhh0, wihL, whhL, bihL, bhhL,
      bng, bnb, bnm, bnv, fcW, fcb, fcWout, fcbout, out);
}
